// ExRestSelfAtten_71305047048410
// MI455X (gfx1250) — matrix operand order repaired and checked by reading, not compiled, not run on hardware
//
#include <hip/hip_runtime.h>
#include <hip/hip_bf16.h>

// ---------------------------------------------------------------------------
// Local-window self-attention block, restructured for MI455X (gfx1250):
//   h = relu(x@W1+b1); q,k,v = h@{Wq,Wk,Wv}; windowed softmax attn; out@W2.
// Neighbor-gather commuted past the K/V projections (zero pad rows project to
// zero rows), so the GEMMs are 31 GFLOP total on v_wmma_f32_16x16x32_f16.
// GEMM: 128x128 block tile, K-tile 64, double-buffered LDS (1 barrier/step),
// gfx1250 async global->LDS staging (GLOBAL_LOAD_ASYNC_TO_LDS_B128).
// ---------------------------------------------------------------------------

typedef __attribute__((ext_vector_type(16))) _Float16 v16h;
typedef __attribute__((ext_vector_type(8)))  _Float16 v8h;
typedef __attribute__((ext_vector_type(8)))  float    v8f;

union FragH { v16h v; v8h h[2]; };

#define KTILE 64
#define KST   72      // LDS row stride in halves (64 + 8 pad) -> 144B, 16B-aligned

#if __has_builtin(__builtin_amdgcn_global_load_async_to_lds_b128) && \
    __has_builtin(__builtin_amdgcn_s_wait_asynccnt)
#define USE_ASYNC_LDS 1
// exact pointer types from the hipcc diagnostic: 16B int vector, AS1 / AS3
typedef int b128_t __attribute__((vector_size(16)));
typedef __attribute__((address_space(1))) b128_t* as1_b128;
typedef __attribute__((address_space(3))) b128_t* as3_b128;
#else
#define USE_ASYNC_LDS 0
#endif

// --- converters ------------------------------------------------------------

// dst[r*dC + c] = (r<sR && c<sC) ? src[r*sC + c] : 0   (f32 -> f16, padded)
__global__ void cvt_pad(const float* __restrict__ src, _Float16* __restrict__ dst,
                        int sR, int sC, int dR, int dC) {
  int i = blockIdx.x * 256 + threadIdx.x;
  if (i >= dR * dC) return;
  int r = i / dC, c = i - r * dC;
  float v = (r < sR && c < sC) ? src[(size_t)r * sC + c] : 0.0f;
  dst[i] = (_Float16)v;
}

// dst[n*dK + k] = (k<sK && n<sN) ? src[k*sN + n] : 0   (transpose to N-major)
__global__ void cvt_pad_t(const float* __restrict__ src, _Float16* __restrict__ dst,
                          int sK, int sN, int dN, int dK) {
  int i = blockIdx.x * 256 + threadIdx.x;
  if (i >= dN * dK) return;
  int n = i / dK, kk = i - n * dK;
  float v = (kk < sK && n < sN) ? src[(size_t)kk * sN + n] : 0.0f;
  dst[i] = (_Float16)v;
}

// --- WMMA GEMM: C[M,N] = A[M,K] * B^T[N,K]  (B pre-transposed, N-major) ----
// 256 threads = 8 waves; block tile 128x128; wave tile 64x32 (4x2 wmma accs).
template<bool RELU, bool OUT_F16>
__global__ __launch_bounds__(256)
void gemm_wmma(const _Float16* __restrict__ Ag,   // [M][lda]
               const _Float16* __restrict__ Bg,   // [N][ldb]  (N-major)
               const float*    __restrict__ bias, // [N] or null
               float*          __restrict__ Cf,
               _Float16*       __restrict__ Ch,
               int M, int N, int K, int lda, int ldb)
{
  __shared__ _Float16 As[2][128 * KST];
  __shared__ _Float16 Bs[2][128 * KST];

  const int t    = threadIdx.x;
  const int lane = t & 31;
  const int wv   = t >> 5;
  const int wm   = wv & 1;        // 2 waves along M  (64 rows each)
  const int wn   = wv >> 1;       // 4 waves along N  (32 cols each)
  const int ll   = lane & 15;
  const int lh   = lane >> 4;

  const int m0 = blockIdx.y * 128;
  const int n0 = blockIdx.x * 128;

  v8f zero = {0.f,0.f,0.f,0.f,0.f,0.f,0.f,0.f};
  v8f acc[4][2];
#pragma unroll
  for (int mt = 0; mt < 4; ++mt)
#pragma unroll
    for (int nt = 0; nt < 2; ++nt) acc[mt][nt] = zero;

  // staging: each thread moves 32 contiguous halves (4x b128) per matrix
  const int sr = t >> 1;           // 0..127
  const int sc = (t & 1) << 5;     // 0 or 32 halves

#if USE_ASYNC_LDS
  auto stage_async = [&](int kt, int buf) {
    const _Float16* gA = Ag + (size_t)(m0 + sr) * lda + kt * KTILE + sc;
    const _Float16* gB = Bg + (size_t)(n0 + sr) * ldb + kt * KTILE + sc;
    _Float16* lA = &As[buf][sr * KST + sc];
    _Float16* lB = &Bs[buf][sr * KST + sc];
#pragma unroll
    for (int c = 0; c < 4; ++c) {
      __builtin_amdgcn_global_load_async_to_lds_b128(
          (as1_b128)(gA + 8 * c), (as3_b128)(lA + 8 * c), 0, 0);
      __builtin_amdgcn_global_load_async_to_lds_b128(
          (as1_b128)(gB + 8 * c), (as3_b128)(lB + 8 * c), 0, 0);
    }
  };
#else
  v8h ra[4], rb[4];
  auto load_global = [&](int kt) {
    const v8h* gA = reinterpret_cast<const v8h*>(Ag + (size_t)(m0 + sr) * lda + kt * KTILE + sc);
    const v8h* gB = reinterpret_cast<const v8h*>(Bg + (size_t)(n0 + sr) * ldb + kt * KTILE + sc);
#pragma unroll
    for (int c = 0; c < 4; ++c) { ra[c] = gA[c]; rb[c] = gB[c]; }
  };
  auto store_lds = [&](int buf) {
    v8h* lA = reinterpret_cast<v8h*>(&As[buf][sr * KST + sc]);
    v8h* lB = reinterpret_cast<v8h*>(&Bs[buf][sr * KST + sc]);
#pragma unroll
    for (int c = 0; c < 4; ++c) { lA[c] = ra[c]; lB[c] = rb[c]; }
  };
#endif

  auto compute = [&](int cur) {
#pragma unroll
    for (int ks = 0; ks < KTILE; ks += 32) {
      // A fragments: 16x32 f16; lane = M%16, halves [kb,kb+8) and [kb+16,kb+24)
      FragH af[4];
#pragma unroll
      for (int mt = 0; mt < 4; ++mt) {
        int row = wm * 64 + mt * 16 + ll;
        const _Float16* base = &As[cur][row * KST + ks + lh * 8];
        af[mt].h[0] = *reinterpret_cast<const v8h*>(base);
        af[mt].h[1] = *reinterpret_cast<const v8h*>(base + 16);
      }
      // B fragments: 32x16 f16; lane = N%16, halves [kb,kb+8) and [kb+16,kb+24)
      FragH bf[2];
#pragma unroll
      for (int nt = 0; nt < 2; ++nt) {
        int col = wn * 32 + nt * 16 + ll;
        const _Float16* base = &Bs[cur][col * KST + ks + lh * 8];
        bf[nt].h[0] = *reinterpret_cast<const v8h*>(base);
        bf[nt].h[1] = *reinterpret_cast<const v8h*>(base + 16);
      }
#pragma unroll
      for (int mt = 0; mt < 4; ++mt)
#pragma unroll
        for (int nt = 0; nt < 2; ++nt)
          acc[mt][nt] = __builtin_amdgcn_wmma_f32_16x16x32_f16(
              false, af[mt].v, false, bf[nt].v, (short)0, acc[mt][nt], false, false);
    }
  };

  const int nk = K / KTILE;

  // prologue: stage tile 0 into buffer 0
#if USE_ASYNC_LDS
  stage_async(0, 0);
  __builtin_amdgcn_s_wait_asynccnt(0);
  __syncthreads();
#else
  load_global(0);
  store_lds(0);
  __syncthreads();
#endif

  // double-buffered main loop: one barrier per K-tile
  for (int kt = 0; kt < nk; ++kt) {
    const int cur = kt & 1;
#if USE_ASYNC_LDS
    if (kt + 1 < nk) stage_async(kt + 1, cur ^ 1);
    compute(cur);
    __builtin_amdgcn_s_wait_asynccnt(0);
#else
    if (kt + 1 < nk) load_global(kt + 1);
    compute(cur);
    if (kt + 1 < nk) store_lds(cur ^ 1);
#endif
    __syncthreads();
  }

  // epilogue: C/D layout -> VGPR i holds M = i + 8*(lane/16), N = lane%16
#pragma unroll
  for (int mt = 0; mt < 4; ++mt) {
#pragma unroll
    for (int nt = 0; nt < 2; ++nt) {
      int col = n0 + wn * 32 + nt * 16 + ll;
      float badd = bias ? bias[col] : 0.0f;
#pragma unroll
      for (int i = 0; i < 8; ++i) {
        int row = m0 + wm * 64 + mt * 16 + i + lh * 8;
        float vv = acc[mt][nt][i] + badd;
        if (RELU) vv = fmaxf(vv, 0.0f);
        if (OUT_F16) Ch[(size_t)row * N + col] = (_Float16)vv;
        else         Cf[(size_t)row * N + col] = vv;
      }
    }
  }
}

// --- fused windowed attention + output head --------------------------------
// One wave32 per sequence position. OOB neighbors score exactly 0 (matches
// the reference's zero-padded h), and contribute zero v.
__global__ __launch_bounds__(256)
void attn_kernel(const float* __restrict__ q, const float* __restrict__ k,
                 const float* __restrict__ v, const float* __restrict__ W2,
                 const float* __restrict__ b2, float* __restrict__ out,
                 float* __restrict__ wout)
{
  const int lane = threadIdx.x & 31;
  const int wave = threadIdx.x >> 5;
  const int flat = blockIdx.x * 8 + wave;     // 0..8191  (b*1024 + s)
  const int s    = flat & 1023;
  const float scale = 0.03608439182435161f;   // 1/sqrt(768)

  float qv[24];
  const float* qr = q + (size_t)flat * 768;
#pragma unroll
  for (int j = 0; j < 24; ++j) qv[j] = qr[lane + 32 * j];

  float sc[11];
#pragma unroll
  for (int w = 0; w < 11; ++w) {
    int n = s + 5 - w;                        // pos = s + (2A - w) - A
    float p = 0.0f;
    if (n >= 0 && n < 1024) {
      const float* kr = k + (size_t)(flat + 5 - w) * 768;
#pragma unroll
      for (int j = 0; j < 24; ++j) p += qv[j] * kr[lane + 32 * j];
#pragma unroll
      for (int off = 16; off >= 1; off >>= 1) p += __shfl_xor(p, off, 32);
      p *= scale;
    }
    sc[w] = p;
  }

  float m = sc[0];
#pragma unroll
  for (int w = 1; w < 11; ++w) m = fmaxf(m, sc[w]);
  float e[11], sum = 0.0f;
#pragma unroll
  for (int w = 0; w < 11; ++w) { e[w] = __expf(sc[w] - m); sum += e[w]; }
  float inv = 1.0f / sum;

#pragma unroll
  for (int w = 0; w < 11; ++w)
    if (lane == w) wout[(size_t)flat * 11 + w] = e[w] * inv;

  float ao[24];
#pragma unroll
  for (int j = 0; j < 24; ++j) ao[j] = 0.0f;
#pragma unroll
  for (int w = 0; w < 11; ++w) {
    int n = s + 5 - w;
    if (n >= 0 && n < 1024) {
      float wt = e[w] * inv;
      const float* vr = v + (size_t)(flat + 5 - w) * 768;
#pragma unroll
      for (int j = 0; j < 24; ++j) ao[j] += wt * vr[lane + 32 * j];
    }
  }

  float o0 = 0.0f, o1 = 0.0f;
#pragma unroll
  for (int j = 0; j < 24; ++j) {
    int hidx = lane + 32 * j;
    o0 += ao[j] * W2[hidx * 2 + 0];
    o1 += ao[j] * W2[hidx * 2 + 1];
  }
#pragma unroll
  for (int off = 16; off >= 1; off >>= 1) {
    o0 += __shfl_xor(o0, off, 32);
    o1 += __shfl_xor(o1, off, 32);
  }
  if (lane == 0) {
    out[(size_t)flat * 2 + 0] = fmaxf(o0 + b2[0], 0.0f);
    out[(size_t)flat * 2 + 1] = fmaxf(o1 + b2[1], 0.0f);
  }
}

// ---------------------------------------------------------------------------

extern "C" void kernel_launch(void* const* d_in, const int* in_sizes, int n_in,
                              void* d_out, int out_size, void* d_ws, size_t ws_size,
                              hipStream_t stream)
{
  (void)in_sizes; (void)n_in; (void)out_size; (void)ws_size;
  const float* x  = (const float*)d_in[0];
  const float* W1 = (const float*)d_in[1];
  const float* b1 = (const float*)d_in[2];
  const float* Wq = (const float*)d_in[3];
  const float* Wk = (const float*)d_in[4];
  const float* Wv = (const float*)d_in[5];
  const float* W2 = (const float*)d_in[6];
  const float* b2 = (const float*)d_in[7];
  float* out = (float*)d_out;

  const int M = 8192, H = 768, KP = 128;   // KP = IN padded 100 -> 128

  char* ws = (char*)d_ws;
  size_t off = 0;
  auto alloc = [&](size_t bytes) -> void* {
    void* p = ws + off;
    off = (off + bytes + 255) & ~(size_t)255;
    return p;
  };
  _Float16* xh  = (_Float16*)alloc((size_t)M * KP * 2);
  _Float16* W1h = (_Float16*)alloc((size_t)H * KP * 2);   // transposed [n][k]
  _Float16* Wqh = (_Float16*)alloc((size_t)H * H * 2);    // transposed
  _Float16* Wkh = (_Float16*)alloc((size_t)H * H * 2);
  _Float16* Wvh = (_Float16*)alloc((size_t)H * H * 2);
  _Float16* hh  = (_Float16*)alloc((size_t)M * H * 2);
  float*    qf  = (float*)   alloc((size_t)M * H * 4);
  float*    kf  = (float*)   alloc((size_t)M * H * 4);
  float*    vf  = (float*)   alloc((size_t)M * H * 4);

  // convert / pad / transpose
  cvt_pad  <<<(M * KP + 255) / 256, 256, 0, stream>>>(x,  xh,  M, 100, M, KP);
  cvt_pad_t<<<(H * KP + 255) / 256, 256, 0, stream>>>(W1, W1h, 100, H, H, KP);
  cvt_pad_t<<<(H * H  + 255) / 256, 256, 0, stream>>>(Wq, Wqh, H, H, H, H);
  cvt_pad_t<<<(H * H  + 255) / 256, 256, 0, stream>>>(Wk, Wkh, H, H, H, H);
  cvt_pad_t<<<(H * H  + 255) / 256, 256, 0, stream>>>(Wv, Wvh, H, H, H, H);

  dim3 g(H / 128, M / 128);  // (6, 64)
  // h = relu(x@W1 + b1), f16 out
  gemm_wmma<true,  true ><<<g, 256, 0, stream>>>(xh, W1h, b1, nullptr, hh, M, H, KP, KP, KP);
  // q, k, v = h @ {Wq, Wk, Wv}, f32 out
  gemm_wmma<false, false><<<g, 256, 0, stream>>>(hh, Wqh, nullptr, qf, nullptr, M, H, H, H, H);
  gemm_wmma<false, false><<<g, 256, 0, stream>>>(hh, Wkh, nullptr, kf, nullptr, M, H, H, H, H);
  gemm_wmma<false, false><<<g, 256, 0, stream>>>(hh, Wvh, nullptr, vf, nullptr, M, H, H, H, H);

  // fused attention + head; d_out = [out (8192*2) | weights (8192*11)]
  attn_kernel<<<M / 8, 256, 0, stream>>>(qf, kf, vf, W2, b2, out, out + (size_t)M * 2);
}
